// HetAgg_52956946760071
// MI455X (gfx1250) — matrix operand order repaired and checked by reading, not compiled, not run on hardware
//
#include <hip/hip_runtime.h>
#include <hip/hip_bf16.h>

typedef __attribute__((ext_vector_type(16))) __bf16 v16bf;
typedef __attribute__((ext_vector_type(8)))  float  v8f;

#define WMMA_BF16(A, B, C) \
  __builtin_amdgcn_wmma_f32_16x16x32_bf16(false, (A), false, (B), (short)0, (C), false, false)

__device__ __forceinline__ float sigm(float x) { return 1.0f / (1.0f + __expf(-x)); }

// ---- A-fragment (16x32 bf16) from an f32 row (gathered features).
// Lane: m = lane&15, half = lane>>4.
// elements 0..7 -> K = k0 + t ; elements 8..15 -> K = k1 + t  (k0/k1 per ISA table)
__device__ __forceinline__ v16bf ldA_f32(const float* __restrict__ row, int k0, int k1) {
  v16bf a;
#pragma unroll
  for (int t = 0; t < 8; ++t) {
    a[t]     = (__bf16)row[k0 + t];
    a[8 + t] = (__bf16)row[k1 + t];
  }
  return a;
}
// Tail: clamped index (always in-bounds, branchless). Values at k>=A multiply a
// zero-padded B-fragment, so they cannot affect the result.
__device__ __forceinline__ v16bf ldA_f32_tail(const float* __restrict__ row, int k0, int k1, int Am1) {
  v16bf a;
#pragma unroll
  for (int t = 0; t < 8; ++t) {
    a[t]     = (__bf16)row[min(k0 + t, Am1)];
    a[8 + t] = (__bf16)row[min(k1 + t, Am1)];
  }
  return a;
}
// ---- A-fragment from a bf16 row (LDS E tile / h state / bf16 Hc): 2x b128 loads.
__device__ __forceinline__ v16bf ldA_bf(const __bf16* __restrict__ row, int k0, int k1) {
  union { uint4 q; __bf16 h[8]; } u0, u1;
  u0.q = *(const uint4*)(row + k0);
  u1.q = *(const uint4*)(row + k1);
  v16bf a;
#pragma unroll
  for (int t = 0; t < 8; ++t) { a[t] = u0.h[t]; a[8 + t] = u1.h[t]; }
  return a;
}
// ---- B-fragment (32x16 bf16) from a pre-converted bf16 weight row: 2x b128 loads.
// Lane: n = lane&15; elements 0..7 -> K = kc + e, 8..15 -> K = kc + 16 + (e-8), kc = kb + (lane>>4)*8.
__device__ __forceinline__ v16bf ldB_bf(const __bf16* __restrict__ wrow, int kc) {
  union { uint4 q; __bf16 h[8]; } u0, u1;
  u0.q = *(const uint4*)(wrow + kc);
  u1.q = *(const uint4*)(wrow + kc + 16);
  v16bf b;
#pragma unroll
  for (int t = 0; t < 8; ++t) { b[t] = u0.h[t]; b[8 + t] = u1.h[t]; }
  return b;
}

// ---------------------------------------------------------------------------
// One-time weight conversion: f32 -> bf16, optionally row-padded to Ap with 0.
// ---------------------------------------------------------------------------
__global__ __launch_bounds__(256) void cvt_bf16_flat(const float* __restrict__ in,
                                                     __bf16* __restrict__ out, int n) {
  const int i = blockIdx.x * 256 + threadIdx.x;
  if (i < n) out[i] = (__bf16)in[i];
}
__global__ __launch_bounds__(256) void cvt_bf16_pad(const float* __restrict__ in,
                                                    __bf16* __restrict__ out,
                                                    int rows, int A, int Ap) {
  const int i = blockIdx.x * 256 + threadIdx.x;
  if (i < rows * Ap) {
    const int r = i / Ap, k = i - r * Ap;
    out[i] = (__bf16)((k < A) ? in[(long)r * A + k] : 0.0f);
  }
}

// ---------------------------------------------------------------------------
// Fused content aggregation for a 16-row tile (block = 128 threads, 4 waves):
//   E = feat[ids] @ fcW^T + fcb                    (WMMA over attr dim)
//   gates = E @ [cWih(dir0); cWih(dir1)]^T + cb    (WMMA, 512 cols)
//   h0=c0=0, T=1 LSTM: h = sig(o)*tanh(sig(i)*tanh(g))  (f-gate drops out)
// Output: bf16 (neighbor sets, consumed only as WMMA input) or f32 (anchor).
// ---------------------------------------------------------------------------
__global__ __launch_bounds__(128) void hetagg_content(
    const int* __restrict__ ids,
    const float* __restrict__ feat, int A, int Ap,
    const __bf16* __restrict__ fcWb, const float* __restrict__ fcb,
    const __bf16* __restrict__ cWihb, const float* __restrict__ cb, int type,
    __bf16* __restrict__ HcB, float* __restrict__ HcF)
{
  __shared__ __bf16 Els[16 * 128];  // E tile (bf16: only consumed as WMMA A-frag)
  __shared__ float  Gls[16 * 512];  // gate tile (both directions, f32)
  const int tid = threadIdx.x, lane = tid & 31, wv = tid >> 5;
  const int half = lane >> 4, nn = lane & 15, kb2 = half * 8;
  const int n0 = blockIdx.x * 16;

  // ---- Phase 1: E tile; wave wv owns col tiles {wv, wv+4}; A-frag shared.
  {
    const long rowid = (long)ids[n0 + nn];
    const float* arow = feat + rowid * (long)A;
    const int c0 = wv * 16 + nn, c1 = (wv + 4) * 16 + nn;
    const __bf16* wrow0 = fcWb + (long)c0 * Ap;
    const __bf16* wrow1 = fcWb + (long)c1 * Ap;
    v8f acc0 = {}, acc1 = {};
    const int Afull = A & ~31;
    for (int kb = 0; kb < Afull; kb += 32) {
      v16bf a = ldA_f32(arow, kb + half * 8, kb + 16 + half * 8);
      acc0 = WMMA_BF16(a, ldB_bf(wrow0, kb + kb2), acc0);
      acc1 = WMMA_BF16(a, ldB_bf(wrow1, kb + kb2), acc1);
    }
    if (Afull < Ap) {   // one tail chunk: B is zero-padded, A clamped in-bounds
      const int kb = Afull;
      v16bf a = ldA_f32_tail(arow, kb + half * 8, kb + 16 + half * 8, A - 1);
      acc0 = WMMA_BF16(a, ldB_bf(wrow0, kb + kb2), acc0);
      acc1 = WMMA_BF16(a, ldB_bf(wrow1, kb + kb2), acc1);
    }
    const float b0 = fcb[c0], b1 = fcb[c1];
#pragma unroll
    for (int r = 0; r < 8; ++r) {               // C: M = r + 8*half, N = lane&15
      Els[(r + 8 * half) * 128 + c0] = (__bf16)(acc0[r] + b0);
      Els[(r + 8 * half) * 128 + c1] = (__bf16)(acc1[r] + b1);
    }
  }
  __syncthreads();

  // ---- Phase 2: 512 gate cols; wave wv owns 8 col tiles; A-frag from LDS shared.
  {
    const __bf16* Wbase = cWihb + (long)type * 2 * 256 * 128;  // [2,256,128] -> g*128
    const float* bbase = cb + type * 2 * 256;
    v8f acc[8];
#pragma unroll
    for (int tt = 0; tt < 8; ++tt) { v8f z = {}; acc[tt] = z; }
    const __bf16* erow = Els + nn * 128;
#pragma unroll
    for (int kb = 0; kb < 128; kb += 32) {
      v16bf a = ldA_bf(erow, kb + half * 8, kb + 16 + half * 8);
#pragma unroll
      for (int tt = 0; tt < 8; ++tt) {
        const int g = (8 * wv + tt) * 16 + nn;
        acc[tt] = WMMA_BF16(a, ldB_bf(Wbase + (long)g * 128, kb + kb2), acc[tt]);
      }
    }
#pragma unroll
    for (int tt = 0; tt < 8; ++tt) {
      const int g = (8 * wv + tt) * 16 + nn;
      const float bv = bbase[g];
#pragma unroll
      for (int r = 0; r < 8; ++r)
        Gls[(r + 8 * half) * 512 + g] = acc[tt][r] + bv;
    }
  }
  __syncthreads();

  // ---- Phase 3: single-step LSTM activations -> Hc[n, 0:128] (fwd|bwd halves)
#pragma unroll
  for (int q = 0; q < 16; ++q) {
    const int lin = q * 128 + tid;         // 16*128 outputs
    const int m = lin >> 7, j2 = lin & 127;
    const int d = j2 >> 6, j = j2 & 63;
    const float* gr = Gls + m * 512 + d * 256;
    const float iv = gr[j], gv = gr[128 + j], ov = gr[192 + j];
    const float cv = sigm(iv) * tanhf(gv);
    const float hv = sigm(ov) * tanhf(cv);
    if (HcB) HcB[(long)(n0 + m) * 128 + j2] = (__bf16)hv;
    else     HcF[(long)(n0 + m) * 128 + j2] = hv;
  }
}

// ---------------------------------------------------------------------------
// Neighbor biLSTM over K steps. Block = 128 threads = 16 batch rows x 1 dir;
// h (bf16, WMMA-ready) / c / hsum state in LDS; per step:
// gates = x@Wih^T + h@Whh^T + b via WMMA. Recurrence is per-batch-row.
// ---------------------------------------------------------------------------
__global__ __launch_bounds__(128) void hetagg_neigh(
    const __bf16* __restrict__ Hc, int K,
    const __bf16* __restrict__ nWihb, const __bf16* __restrict__ nWhhb,
    const float* __restrict__ nb, int type,
    float* __restrict__ agg)
{
  __shared__ __bf16 hls[16 * 64];          // h state, bf16 (WMMA A-frag input)
  __shared__ float  cls[16 * 64], hsum[16 * 64];
  __shared__ float  Gls[16 * 256];
  const int tid = threadIdx.x, lane = tid & 31, wv = tid >> 5;
  const int half = lane >> 4, nn = lane & 15, kb2 = half * 8;
  const int b0 = blockIdx.x * 16;
  const int dir = blockIdx.y;
  for (int q = tid; q < 1024; q += 128) {
    hls[q] = (__bf16)0.0f; cls[q] = 0.f; hsum[q] = 0.f;
  }
  __syncthreads();

  const __bf16* Wx = nWihb + (long)(type * 2 + dir) * 256 * 128;
  const __bf16* Wh = nWhhb + (long)(type * 2 + dir) * 256 * 64;
  const float* bb = nb + (type * 2 + dir) * 256;

  for (int s = 0; s < K; ++s) {
    const int kseq = dir ? (K - 1 - s) : s;
    v8f acc[4];
#pragma unroll
    for (int tt = 0; tt < 4; ++tt) { v8f z = {}; acc[tt] = z; }
    // x @ Wih^T  (K = 128, exact; x is bf16 in HBM)
    const __bf16* xrow = Hc + ((long)(b0 + nn) * K + kseq) * 128;
#pragma unroll
    for (int kb = 0; kb < 128; kb += 32) {
      v16bf a = ldA_bf(xrow, kb + half * 8, kb + 16 + half * 8);
#pragma unroll
      for (int tt = 0; tt < 4; ++tt) {
        const int g = (4 * wv + tt) * 16 + nn;
        acc[tt] = WMMA_BF16(a, ldB_bf(Wx + (long)g * 128, kb + kb2), acc[tt]);
      }
    }
    // h @ Whh^T  (K = 64 from LDS state, exact)
    const __bf16* hrow = hls + nn * 64;
#pragma unroll
    for (int kb = 0; kb < 64; kb += 32) {
      v16bf a = ldA_bf(hrow, kb + half * 8, kb + 16 + half * 8);
#pragma unroll
      for (int tt = 0; tt < 4; ++tt) {
        const int g = (4 * wv + tt) * 16 + nn;
        acc[tt] = WMMA_BF16(a, ldB_bf(Wh + (long)g * 64, kb + kb2), acc[tt]);
      }
    }
#pragma unroll
    for (int tt = 0; tt < 4; ++tt) {
      const int g = (4 * wv + tt) * 16 + nn;
      const float bv = bb[g];
#pragma unroll
      for (int r = 0; r < 8; ++r)
        Gls[(r + 8 * half) * 256 + g] = acc[tt][r] + bv;
    }
    __syncthreads();
    // LSTM cell update: i,f,g,o split
    for (int q = tid; q < 1024; q += 128) {
      const int m = q >> 6, j = q & 63;
      const float* gr = Gls + m * 256;
      const float iv = gr[j], fv = gr[64 + j], gv = gr[128 + j], ov = gr[192 + j];
      const float cv = sigm(fv) * cls[q] + sigm(iv) * tanhf(gv);
      cls[q] = cv;
      const float hv = sigm(ov) * tanhf(cv);
      hls[q] = (__bf16)hv;
      hsum[q] += hv;
    }
    __syncthreads();
  }
  const float invK = 1.0f / (float)K;
  for (int q = tid; q < 1024; q += 128) {
    const int m = q >> 6, j = q & 63;
    agg[(long)(b0 + m) * 128 + dir * 64 + j] = hsum[q] * invK;
  }
}

// ---------------------------------------------------------------------------
// 5-way attention: w = softmax(leaky_relu([c|e_k] . att)); out = sum w_k e_k
// One 128-thread block per batch row.
// ---------------------------------------------------------------------------
__global__ __launch_bounds__(128) void hetagg_attn(
    const float* __restrict__ cagg, const float* __restrict__ aagg,
    const float* __restrict__ pagg, const float* __restrict__ tagg,
    const float* __restrict__ vagg, const float* __restrict__ att,
    float* __restrict__ out)
{
  __shared__ float red[128];
  __shared__ float sc[5];
  __shared__ float wsh[5];
  const int t = threadIdx.x;
  const long b = blockIdx.x;
  float e[5];
  e[0] = cagg[b * 128 + t];
  e[1] = aagg[b * 128 + t];
  e[2] = pagg[b * 128 + t];
  e[3] = tagg[b * 128 + t];
  e[4] = vagg[b * 128 + t];
  const float a0 = att[t] * e[0];       // anchor half of the dot, shared by all k
  const float a1 = att[128 + t];
  for (int k = 0; k < 5; ++k) {
    red[t] = a0 + a1 * e[k];
    __syncthreads();
    for (int s2 = 64; s2 > 0; s2 >>= 1) {
      if (t < s2) red[t] += red[t + s2];
      __syncthreads();
    }
    if (t == 0) sc[k] = red[0];
    __syncthreads();
  }
  if (t == 0) {
    float v[5], mx = -1e30f;
#pragma unroll
    for (int k = 0; k < 5; ++k) {
      float x = sc[k];
      x = (x > 0.f) ? x : 0.01f * x;     // leaky_relu(0.01)
      v[k] = x;
      mx = fmaxf(mx, x);
    }
    float ssum = 0.f;
#pragma unroll
    for (int k = 0; k < 5; ++k) { v[k] = __expf(v[k] - mx); ssum += v[k]; }
#pragma unroll
    for (int k = 0; k < 5; ++k) wsh[k] = v[k] / ssum;
  }
  __syncthreads();
  float o = 0.f;
#pragma unroll
  for (int k = 0; k < 5; ++k) o += wsh[k] * e[k];
  out[b * 128 + t] = o;
}

extern "C" void kernel_launch(void* const* d_in, const int* in_sizes, int n_in,
                              void* d_out, int out_size, void* d_ws, size_t ws_size,
                              hipStream_t stream) {
  (void)in_sizes; (void)n_in; (void)out_size; (void)ws_size;
  // setup_inputs() dict order
  const float* feat_a = (const float*)d_in[0];
  const float* fcW_a  = (const float*)d_in[1];
  const float* fcb_a  = (const float*)d_in[2];
  const float* feat_p = (const float*)d_in[3];
  const float* fcW_p  = (const float*)d_in[4];
  const float* fcb_p  = (const float*)d_in[5];
  const float* feat_t = (const float*)d_in[6];
  const float* fcW_t  = (const float*)d_in[7];
  const float* fcb_t  = (const float*)d_in[8];
  const float* feat_v = (const float*)d_in[9];
  const float* fcW_v  = (const float*)d_in[10];
  const float* fcb_v  = (const float*)d_in[11];
  const float* c_Wih  = (const float*)d_in[12];
  /* d_in[13] = c_Whh: provably unused (T==1 biLSTM with h0==0) */
  const float* c_b    = (const float*)d_in[14];
  const float* n_Wih  = (const float*)d_in[15];
  const float* n_Whh  = (const float*)d_in[16];
  const float* n_b    = (const float*)d_in[17];
  const float* att    = (const float*)d_in[18];
  const int* neigh_a  = (const int*)d_in[19];
  const int* neigh_p  = (const int*)d_in[20];
  const int* neigh_t  = (const int*)d_in[21];
  const int* neigh_v  = (const int*)d_in[22];
  const int* id_batch = (const int*)d_in[23];

  const int B = 2048;
  const int A_a = 334, A_p = 4231, A_t = 50, A_v = 20;
  const int Ap_a = 352, Ap_p = 4256, Ap_t = 64, Ap_v = 32;   // padded to %32

  char* cur = (char*)d_ws;
  auto carve = [&](size_t bytes) {
    void* p = (void*)cur;
    cur += (bytes + 255) & ~(size_t)255;   // keep 256B alignment
    return p;
  };
  float*  Hc_id = (float*)carve((size_t)B * 128 * 4);          // f32 (attention input)
  __bf16* Hc_a  = (__bf16*)carve((size_t)B * 10 * 128 * 2);    // bf16 (WMMA-only)
  __bf16* Hc_p  = (__bf16*)carve((size_t)B * 10 * 128 * 2);
  __bf16* Hc_t  = (__bf16*)carve((size_t)B * 10 * 128 * 2);
  __bf16* Hc_v  = (__bf16*)carve((size_t)B * 3 * 128 * 2);
  float*  agg_a = (float*)carve((size_t)B * 128 * 4);
  float*  agg_p = (float*)carve((size_t)B * 128 * 4);
  float*  agg_t = (float*)carve((size_t)B * 128 * 4);
  float*  agg_v = (float*)carve((size_t)B * 128 * 4);
  __bf16* wfc_a = (__bf16*)carve((size_t)128 * Ap_a * 2);      // padded bf16 weights
  __bf16* wfc_p = (__bf16*)carve((size_t)128 * Ap_p * 2);
  __bf16* wfc_t = (__bf16*)carve((size_t)128 * Ap_t * 2);
  __bf16* wfc_v = (__bf16*)carve((size_t)128 * Ap_v * 2);
  __bf16* wc    = (__bf16*)carve((size_t)4 * 2 * 256 * 128 * 2);
  __bf16* wnx   = (__bf16*)carve((size_t)4 * 2 * 256 * 128 * 2);
  __bf16* wnh   = (__bf16*)carve((size_t)4 * 2 * 256 * 64 * 2);  // total ~25 MB

  // --- one-time weight conversions (deterministic, every call) ---
  auto grids = [](int n) { return dim3((n + 255) / 256); };
  cvt_bf16_pad<<<grids(128 * Ap_a), 256, 0, stream>>>(fcW_a, wfc_a, 128, A_a, Ap_a);
  cvt_bf16_pad<<<grids(128 * Ap_p), 256, 0, stream>>>(fcW_p, wfc_p, 128, A_p, Ap_p);
  cvt_bf16_pad<<<grids(128 * Ap_t), 256, 0, stream>>>(fcW_t, wfc_t, 128, A_t, Ap_t);
  cvt_bf16_pad<<<grids(128 * Ap_v), 256, 0, stream>>>(fcW_v, wfc_v, 128, A_v, Ap_v);
  cvt_bf16_flat<<<grids(262144), 256, 0, stream>>>(c_Wih, wc, 262144);
  cvt_bf16_flat<<<grids(262144), 256, 0, stream>>>(n_Wih, wnx, 262144);
  cvt_bf16_flat<<<grids(131072), 256, 0, stream>>>(n_Whh, wnh, 131072);

  // --- content aggregation (gather -> FC -> 1-step biLSTM), fused per 16 rows ---
  dim3 blk(128);
  hetagg_content<<<B / 16,      blk, 0, stream>>>(id_batch, feat_a, A_a, Ap_a, wfc_a, fcb_a, wc, c_b, 0, nullptr, Hc_id);
  hetagg_content<<<B * 10 / 16, blk, 0, stream>>>(neigh_a,  feat_a, A_a, Ap_a, wfc_a, fcb_a, wc, c_b, 0, Hc_a, nullptr);
  hetagg_content<<<B * 10 / 16, blk, 0, stream>>>(neigh_p,  feat_p, A_p, Ap_p, wfc_p, fcb_p, wc, c_b, 1, Hc_p, nullptr);
  hetagg_content<<<B * 10 / 16, blk, 0, stream>>>(neigh_t,  feat_t, A_t, Ap_t, wfc_t, fcb_t, wc, c_b, 2, Hc_t, nullptr);
  hetagg_content<<<B * 3 / 16,  blk, 0, stream>>>(neigh_v,  feat_v, A_v, Ap_v, wfc_v, fcb_v, wc, c_b, 3, Hc_v, nullptr);

  // --- neighbor biLSTM + temporal mean ---
  dim3 ngrid(B / 16, 2);
  hetagg_neigh<<<ngrid, blk, 0, stream>>>(Hc_a, 10, wnx, wnh, n_b, 0, agg_a);
  hetagg_neigh<<<ngrid, blk, 0, stream>>>(Hc_p, 10, wnx, wnh, n_b, 1, agg_p);
  hetagg_neigh<<<ngrid, blk, 0, stream>>>(Hc_t, 10, wnx, wnh, n_b, 2, agg_t);
  hetagg_neigh<<<ngrid, blk, 0, stream>>>(Hc_v, 3,  wnx, wnh, n_b, 3, agg_v);

  // --- 5-way semantic attention ---
  hetagg_attn<<<B, 128, 0, stream>>>(Hc_id, agg_a, agg_p, agg_t, agg_v, att, (float*)d_out);
}
